// L5_16887811408602
// MI455X (gfx1250) — compile-verified, matrix operand order checked by reading, not run on hardware
//
#include <hip/hip_runtime.h>

// MI455X / gfx1250 fused kernel:
//   out = relu( (mask ? 0 : x) @ W + b + 0.3 * l23 )
// B=32768, D_IN=2048, D_LAT=64. HBM-bound streaming x+mask (~336 MB -> ~14.5us
// floor at 23.3 TB/s). Matmul via v_wmma_f32_16x16x32_bf16 (wave32 WMMA).
// W is pre-swizzled ONCE into the exact WMMA B-fragment bf16 layout in d_ws,
// then each 16 KB K-chunk is DMA'd to LDS by the Tensor Data Mover
// (tensor_load_to_lds, TENSORcnt) with double buffering.
// Mask application is branchless: v_perm_b32 gathers the 0/1 bool bytes,
// (pr<<16)-pr expands them to 16-bit zero-masks, and-not applies them to the
// packed bf16 pairs. Conversion uses native __bf16 casts (v_cvt_pk_bf16_f32).

typedef __attribute__((ext_vector_type(8)))  float        v8f;
typedef __attribute__((ext_vector_type(16))) __bf16       v16bf;
typedef __attribute__((ext_vector_type(8)))  __bf16       v8bf;
typedef __attribute__((ext_vector_type(4)))  unsigned int u32x4;
typedef __attribute__((ext_vector_type(8)))  int          i32x8;
typedef __attribute__((ext_vector_type(4)))  int          i32x4;

#define B_ROWS  32768
#define D_IN    2048
#define D_LAT   64
#define MODF    0.3f

#define KC      128                    // K chunk staged in LDS per iteration
#define NCHUNK  (D_IN / KC)            // 16
#define WG_M    128                    // output rows per workgroup (8 waves x 16)
#define THREADS 256
#define CHUNK_BYTES (KC * D_LAT * 2)   // 16 KB of bf16 fragments per chunk
#define WSWZ_BYTES  (D_IN * D_LAT * 2) // 256 KB swizzled W

__device__ __forceinline__ unsigned short f32_to_bf16(float f) {
  unsigned int u = __builtin_bit_cast(unsigned int, f);
  u += 0x7fffu + ((u >> 16) & 1u);
  return (unsigned short)(u >> 16);
}

union U16x8 { unsigned short s[8]; uint4 q; };
union APack { v16bf v; __bf16 e[16]; unsigned u[8]; };

// Build the 16-element bf16 A fragment from 16 f32 values and 16 mask bytes
// (packed in 4 dwords, bytes are 0/1). Branchless:
//   pr = mb0 | mb1<<16      (one v_perm_b32)
//   zmask = (pr<<16) - pr   (= mb0*0xFFFF | mb1*0xFFFF0000, no carry)
//   dword &= ~zmask
__device__ __forceinline__ void build_a_masked(const float xf[16],
                                               const unsigned mw[4], APack& a) {
  #pragma unroll
  for (int i = 0; i < 16; ++i) a.e[i] = (__bf16)xf[i];   // native cvt (RNE)
  #pragma unroll
  for (int p = 0; p < 8; ++p) {
    const unsigned w   = mw[p >> 1];
    const unsigned sel = (p & 1) ? 0x0C030C02u : 0x0C010C00u; // bytes j1,j0 of w
    const unsigned pr  = __builtin_amdgcn_perm(0u, w, sel);   // mb0 | mb1<<16
    a.u[p] &= ~((pr << 16) - pr);
  }
}

// ---------------------------------------------------------------------------
// Prep kernel: convert W[2048x64] f32 -> bf16, pre-swizzled into the WMMA
// 16-bit B-matrix lane layout, chunk-contiguous so each 16 KB chunk is a flat
// LDS image. Fragment (c,s,nt): lane L holds column n = nt*16+(L&15),
// k = h*8+{0..7} (at lane*16) and 16+h*8+{0..7} (at +512), h = L>>4.
// ---------------------------------------------------------------------------
__global__ __launch_bounds__(THREADS)
void swizzle_w_bf16(const float* __restrict__ W, unsigned char* __restrict__ wswz) {
  const int e  = blockIdx.x * THREADS + threadIdx.x;  // 0..8191
  const int c  = e >> 9;
  const int s  = (e >> 7) & 3;
  const int nt = (e >> 5) & 3;
  const int ln = e & 31;
  const int hh = ln >> 4;
  const int n  = nt * 16 + (ln & 15);
  const int k0 = c * KC + s * 32 + hh * 8;
  U16x8 lo, hi;
  #pragma unroll
  for (int j = 0; j < 8; ++j) {
    lo.s[j] = f32_to_bf16(W[(size_t)(k0 + j)      * D_LAT + n]);
    hi.s[j] = f32_to_bf16(W[(size_t)(k0 + 16 + j) * D_LAT + n]);
  }
  unsigned char* dst = wswz + (size_t)c * CHUNK_BYTES + (s * 4 + nt) * 1024 + ln * 16;
  *reinterpret_cast<uint4*>(dst)       = lo.q;
  *reinterpret_cast<uint4*>(dst + 512) = hi.q;
}

// ---------------------------------------------------------------------------
// TDM: issue tensor_load_to_lds for one 16 KB chunk (1D tile: 2048 x 8-byte
// elements). D# built per CDNA5 ISA ch.8. One instruction per issuing wave.
// ---------------------------------------------------------------------------
__device__ __forceinline__ void tdm_load_chunk(const unsigned char* gsrc,
                                               unsigned int lds_addr) {
#if defined(__AMDGCN__) && __has_builtin(__builtin_amdgcn_tensor_load_to_lds)
  const unsigned long long ga = (unsigned long long)(size_t)gsrc;
  u32x4 g0;
  g0[0] = 1u;                                  // count=1, user descriptor
  g0[1] = lds_addr;                            // lds_addr[31:0]
  g0[2] = (unsigned int)(ga & 0xffffffffull);  // global_addr[31:0]
  g0[3] = (unsigned int)((ga >> 32) & 0x01ffffffull) | (2u << 30); // [56:32] | type=2

  i32x8 g1;
  g1[0] = (int)(3u << 16);                       // wg_mask=0, data_size=8B
  g1[1] = (int)(((WSWZ_BYTES / 8) & 0xffffu) << 16); // tensor_dim0[15:0] @ bits63:48
  g1[2] = (int)(((WSWZ_BYTES / 8) >> 16) & 0xffffu)  // tensor_dim0[31:16]
        | (int)(1u << 16);                       // tensor_dim1 = 1 (bits 95:80)
  g1[3] = (int)(((unsigned)(CHUNK_BYTES / 8)) << 16); // tile_dim0 = 2048 @ bits127:112
  g1[4] = 1;                                     // tile_dim1 = 1, tile_dim2 = 0
  g1[5] = (int)(WSWZ_BYTES / 8);                 // tensor_dim0_stride[31:0]
  g1[6] = 0;                                     // stride hi, dim1_stride lo
  g1[7] = 0;

  i32x4 gz = {0, 0, 0, 0};
#if defined(__clang_major__) && (__clang_major__ >= 23)
  i32x8 gz8 = {0, 0, 0, 0, 0, 0, 0, 0};
  __builtin_amdgcn_tensor_load_to_lds(g0, g1, gz, gz, gz8, 0);
#else
  __builtin_amdgcn_tensor_load_to_lds(g0, g1, gz, gz, 0);
#endif
#else
  (void)gsrc; (void)lds_addr;
#endif
}

__device__ __forceinline__ void tdm_wait0() {
#if defined(__AMDGCN__) && __has_builtin(__builtin_amdgcn_s_wait_tensorcnt)
  __builtin_amdgcn_s_wait_tensorcnt(0);
#endif
}

#if defined(__AMDGCN__) && __has_builtin(__builtin_amdgcn_tensor_load_to_lds)
#define TDM_AVAILABLE 1
#else
#define TDM_AVAILABLE 0
#endif

// ---------------------------------------------------------------------------
// Shared per-chunk compute: 4 K-substeps x (A fragment from global, 4 B
// fragments from LDS, 4 WMMAs). Branchless throughout (EXEC stays all-ones).
// ---------------------------------------------------------------------------
__device__ __forceinline__ void compute_chunk(const float* __restrict__ xrow,
                                              const unsigned char* __restrict__ mrow,
                                              const unsigned char* __restrict__ buf,
                                              int kc, int h, int lane, v8f acc[4]) {
  #pragma unroll
  for (int s = 0; s < 4; ++s) {
    const int kb0 = kc + s * 32 + h * 8;
    __builtin_prefetch(xrow + kb0 + KC, 0, 3);   // global_prefetch_b8

    const float4 f0 = *reinterpret_cast<const float4*>(xrow + kb0);
    const float4 f1 = *reinterpret_cast<const float4*>(xrow + kb0 + 4);
    const float4 f2 = *reinterpret_cast<const float4*>(xrow + kb0 + 16);
    const float4 f3 = *reinterpret_cast<const float4*>(xrow + kb0 + 20);
    const uint2 mq0 = *reinterpret_cast<const uint2*>(mrow + kb0);       // bytes 0..7
    const uint2 mq1 = *reinterpret_cast<const uint2*>(mrow + kb0 + 16);  // bytes 8..15

    const float    xf[16] = {f0.x, f0.y, f0.z, f0.w, f1.x, f1.y, f1.z, f1.w,
                             f2.x, f2.y, f2.z, f2.w, f3.x, f3.y, f3.z, f3.w};
    const unsigned mw[4]  = {mq0.x, mq0.y, mq1.x, mq1.y};

    APack a;
    build_a_masked(xf, mw, a);

    // Preload all four B fragments (8x ds_load_b128), then 4 back-to-back WMMAs.
    v16bf bm[4];
    #pragma unroll
    for (int nt = 0; nt < 4; ++nt) {
      const int fb = (s * 4 + nt) * 1024;
      const v8bf blo = *reinterpret_cast<const v8bf*>(buf + fb +       lane * 16);
      const v8bf bhi = *reinterpret_cast<const v8bf*>(buf + fb + 512 + lane * 16);
      bm[nt] = __builtin_shufflevector(
          blo, bhi, 0, 1, 2, 3, 4, 5, 6, 7, 8, 9, 10, 11, 12, 13, 14, 15);
    }
    #pragma unroll
    for (int nt = 0; nt < 4; ++nt) {
      acc[nt] = __builtin_amdgcn_wmma_f32_16x16x32_bf16(
          false, a.v, false, bm[nt], (short)0, acc[nt], false, false);
    }
  }
}

__device__ __forceinline__ void epilogue(const float* __restrict__ l23,
                                         const float* __restrict__ bias,
                                         float* __restrict__ out,
                                         int mBase, int h, int l16, v8f acc[4]) {
  #pragma unroll
  for (int nt = 0; nt < 4; ++nt) {
    const int col = nt * 16 + l16;
    const float bv = bias[col];
    #pragma unroll
    for (int r = 0; r < 8; ++r) {
      const size_t row = (size_t)(mBase + h * 8 + r);
      float v = acc[nt][r] + bv + MODF * l23[row * D_LAT + col];
      out[row * D_LAT + col] = v > 0.0f ? v : 0.0f;
    }
  }
}

// ---------------------------------------------------------------------------
// Main kernel (TDM variant): W fragments arrive pre-swizzled via d_ws.
// ---------------------------------------------------------------------------
__global__ __launch_bounds__(THREADS)
void fused_thal_fc1_relu_tdm(const float* __restrict__ x,
                             const float* __restrict__ l23,
                             const float* __restrict__ bias,
                             const unsigned char* __restrict__ mask,
                             const unsigned char* __restrict__ wswz,
                             float* __restrict__ out) {
  __shared__ __align__(128) unsigned char lds[2][CHUNK_BYTES];  // double buffer

  const int tid   = threadIdx.x;
  const int lane  = tid & 31;
  const int wave  = tid >> 5;
  const int h     = lane >> 4;
  const int l16   = lane & 15;
  const int mBase = blockIdx.x * WG_M + wave * 16;

  const float* __restrict__ xrow         = x    + (size_t)(mBase + l16) * D_IN;
  const unsigned char* __restrict__ mrow = mask + (size_t)(mBase + l16) * D_IN;

  v8f acc[4] = {};

#if TDM_AVAILABLE
  if (wave == 0) {
    tdm_load_chunk(wswz, (unsigned int)(size_t)(void*)&lds[0][0]);
    tdm_wait0();
  }
#else
  for (int i = tid; i < CHUNK_BYTES / 16; i += THREADS)
    reinterpret_cast<uint4*>(&lds[0][0])[i] =
        reinterpret_cast<const uint4*>(wswz)[i];
#endif
  __syncthreads();

  for (int c = 0; c < NCHUNK; ++c) {
    if (c + 1 < NCHUNK) {
#if TDM_AVAILABLE
      if (wave == 0)
        tdm_load_chunk(wswz + (size_t)(c + 1) * CHUNK_BYTES,
                       (unsigned int)(size_t)(void*)&lds[(c + 1) & 1][0]);
#else
      const uint4* src = reinterpret_cast<const uint4*>(wswz + (size_t)(c + 1) * CHUNK_BYTES);
      uint4* dst = reinterpret_cast<uint4*>(&lds[(c + 1) & 1][0]);
      for (int i = tid; i < CHUNK_BYTES / 16; i += THREADS) dst[i] = src[i];
#endif
    }
    compute_chunk(xrow, mrow, lds[c & 1], c * KC, h, lane, acc);
#if TDM_AVAILABLE
    if (wave == 0) tdm_wait0();   // next chunk's DMA complete before barrier
#endif
    __syncthreads();
  }

  epilogue(l23, bias, out, mBase, h, l16, acc);
}

// ---------------------------------------------------------------------------
// Fallback self-contained kernel (no workspace): inline W staging.
// ---------------------------------------------------------------------------
__device__ __forceinline__ void stage_w(const float* __restrict__ W, int kcBase,
                                        unsigned char* __restrict__ buf, int tid) {
  #pragma unroll
  for (int it = 0; it < 2; ++it) {
    const int e  = tid + it * THREADS;
    const int s  = e >> 7;
    const int nt = (e >> 5) & 3;
    const int ln = e & 31;
    const int hh = ln >> 4;
    const int n  = nt * 16 + (ln & 15);
    const int k0 = kcBase + s * 32 + hh * 8;
    U16x8 lo, hi;
    #pragma unroll
    for (int j = 0; j < 8; ++j) {
      lo.s[j] = f32_to_bf16(W[(size_t)(k0 + j)      * D_LAT + n]);
      hi.s[j] = f32_to_bf16(W[(size_t)(k0 + 16 + j) * D_LAT + n]);
    }
    const int fragBase = (s * 4 + nt) * 1024;
    *reinterpret_cast<uint4*>(buf + fragBase +       ln * 16) = lo.q;
    *reinterpret_cast<uint4*>(buf + fragBase + 512 + ln * 16) = hi.q;
  }
}

__global__ __launch_bounds__(THREADS)
void fused_thal_fc1_relu(const float* __restrict__ x,
                         const float* __restrict__ l23,
                         const float* __restrict__ W,
                         const float* __restrict__ bias,
                         const unsigned char* __restrict__ mask,
                         float* __restrict__ out) {
  __shared__ __align__(16) unsigned char lds[2][CHUNK_BYTES];

  const int tid   = threadIdx.x;
  const int lane  = tid & 31;
  const int wave  = tid >> 5;
  const int h     = lane >> 4;
  const int l16   = lane & 15;
  const int mBase = blockIdx.x * WG_M + wave * 16;

  const float* __restrict__ xrow         = x    + (size_t)(mBase + l16) * D_IN;
  const unsigned char* __restrict__ mrow = mask + (size_t)(mBase + l16) * D_IN;

  v8f acc[4] = {};

  stage_w(W, 0, lds[0], tid);
  __syncthreads();

  for (int c = 0; c < NCHUNK; ++c) {
    if (c + 1 < NCHUNK) stage_w(W, c * KC + KC, lds[(c + 1) & 1], tid);
    compute_chunk(xrow, mrow, lds[c & 1], c * KC, h, lane, acc);
    __syncthreads();
  }

  epilogue(l23, bias, out, mBase, h, l16, acc);
}

extern "C" void kernel_launch(void* const* d_in, const int* in_sizes, int n_in,
                              void* d_out, int out_size, void* d_ws, size_t ws_size,
                              hipStream_t stream) {
  // setup_inputs() order: x, l23_pred, W, b, mask
  const float*         x    = (const float*)d_in[0];
  const float*         l23  = (const float*)d_in[1];
  const float*         W    = (const float*)d_in[2];
  const float*         b    = (const float*)d_in[3];
  const unsigned char* mask = (const unsigned char*)d_in[4];  // jnp bool = 1 byte
  float* out = (float*)d_out;

  dim3 grid(B_ROWS / WG_M);   // 256 workgroups
  dim3 block(THREADS);        // 8 wave32 waves

  if (d_ws != nullptr && ws_size >= (size_t)WSWZ_BYTES) {
    unsigned char* wswz = (unsigned char*)d_ws;
    swizzle_w_bf16<<<dim3(8192 / THREADS), block, 0, stream>>>(W, wswz);
    fused_thal_fc1_relu_tdm<<<grid, block, 0, stream>>>(x, l23, b, mask, wswz, out);
  } else {
    fused_thal_fc1_relu<<<grid, block, 0, stream>>>(x, l23, W, b, mask, out);
  }
}
